// MultiHeadCausalSelfAttention_73615739453934
// MI455X (gfx1250) — matrix operand order repaired and checked by reading, not compiled, not run on hardware
//
#include <hip/hip_runtime.h>
#include <hip/hip_bf16.h>
#include <math.h>

// MultiHeadCausalSelfAttention for MI455X (gfx1250, wave32, WMMA bf16).
// B=4, T=2048, D_MODEL=1024, H=16, DK=64. fp32 in/out, bf16 WMMA w/ f32 accum.
//
// Pipeline:
//   1) convert_x:  x fp32 -> bf16 (row-major)
//   2) convert_wt: Wq/Wk/Wv fp32 -> bf16, transposed to n-major [3][n][k]
//   3) qkv_gemm:   fused bf16 GEMM, async double-buffered LDS staging, WMMA,
//                  bias + 1/sqrt(dk) folded into Q, writes [B,H,T,DK] bf16
//   4) attn:       causal flash attention, async K/V staging, WMMA scores +
//                  WMMA P*V, online softmax in f32, fp32 output

#define D_MODELC 1024
#define T_SEQ    2048
#define BATCH    4
#define NHEADS   16
#define DKC      64

typedef __bf16 bf16_t;
typedef __attribute__((ext_vector_type(16))) __bf16 v16bf;
typedef __attribute__((ext_vector_type(8)))  __bf16 v8bf;
typedef __attribute__((ext_vector_type(4)))  __bf16 v4bf;
typedef __attribute__((ext_vector_type(8)))  float  v8f;
typedef __attribute__((ext_vector_type(4)))  int    v4i;

#if defined(__AMDGCN__) && __has_builtin(__builtin_amdgcn_global_load_async_to_lds_b128)
#define HAS_ASYNC 1
#else
#define HAS_ASYNC 0
#endif

// 16-byte global -> LDS copy (async on CDNA5, sync fallback otherwise).
__device__ __forceinline__ void cp16_g2s(void* lds, const void* glob) {
#if HAS_ASYNC
  __builtin_amdgcn_global_load_async_to_lds_b128((v4i*)glob, (v4i*)lds, 0, 0);
#else
  *(uint4*)lds = *(const uint4*)glob;
#endif
}

__device__ __forceinline__ void wait_async0() {
#if HAS_ASYNC
#if __has_builtin(__builtin_amdgcn_s_wait_asynccnt)
  __builtin_amdgcn_s_wait_asynccnt(0);
#else
  asm volatile("s_wait_asynccnt 0x0" ::: "memory");
#endif
#endif
}

__device__ __forceinline__ v8f wmma_bf16(v16bf a, v16bf b, v8f c) {
  // D = A(16x32 bf16) * B(32x16 bf16) + C(16x16 f32)
  return __builtin_amdgcn_wmma_f32_16x16x32_bf16(false, a, false, b, (short)0, c,
                                                 false, false);
}

__device__ __forceinline__ v16bf join16(v8bf lo, v8bf hi) {
  return __builtin_shufflevector(lo, hi, 0, 1, 2, 3, 4, 5, 6, 7, 8, 9, 10, 11,
                                 12, 13, 14, 15);
}

// A-fragment from a row-major 32-wide row: one contiguous 16-element run.
__device__ __forceinline__ v16bf load_afrag(const bf16_t* row, int kb /*lh*8*/) {
  return join16(*(const v8bf*)(row + 2 * kb), *(const v8bf*)(row + 2 * kb + 8));
}
// B-fragment from an n-major (contraction-contiguous) row: one 16-element run.
__device__ __forceinline__ v16bf load_bfrag(const bf16_t* row, int kbase /*lh*16*/) {
  return join16(*(const v8bf*)(row + kbase), *(const v8bf*)(row + kbase + 8));
}

// ---------------------------------------------------------------------------
// Kernel 1: fp32 -> bf16 (row-major, 4 elements/thread).
// ---------------------------------------------------------------------------
__global__ __launch_bounds__(256)
void convert_x_kernel(const float* __restrict__ in, bf16_t* __restrict__ out,
                      int n4) {
  const int i = blockIdx.x * 256 + threadIdx.x;
  if (i < n4) {
    const float4 f = ((const float4*)in)[i];
    v4bf o = {(bf16_t)f.x, (bf16_t)f.y, (bf16_t)f.z, (bf16_t)f.w};
    ((v4bf*)out)[i] = o;
  }
}

// ---------------------------------------------------------------------------
// Kernel 2: W fp32 [k][n] -> bf16 transposed [n][k] (LDS-tiled, 32x32/block).
// blockIdx.z selects Wq/Wk/Wv.
// ---------------------------------------------------------------------------
__global__ __launch_bounds__(256)
void convert_wt_kernel(const float* __restrict__ W0, const float* __restrict__ W1,
                       const float* __restrict__ W2, bf16_t* __restrict__ outbase) {
  __shared__ float tile[32][33];
  const float* W = (blockIdx.z == 0) ? W0 : (blockIdx.z == 1) ? W1 : W2;
  bf16_t* out = outbase + (size_t)blockIdx.z * D_MODELC * D_MODELC;
  const int k0 = blockIdx.y * 32, n0 = blockIdx.x * 32;
  const int c = threadIdx.x & 31, r0 = threadIdx.x >> 5;
#pragma unroll
  for (int i = 0; i < 4; ++i) {
    const int r = r0 + i * 8;
    tile[r][c] = W[(size_t)(k0 + r) * D_MODELC + n0 + c];
  }
  __syncthreads();
#pragma unroll
  for (int i = 0; i < 4; ++i) {
    const int r = r0 + i * 8;  // n-local
    out[(size_t)(n0 + r) * D_MODELC + k0 + c] = (bf16_t)tile[c][r];
  }
}

// ---------------------------------------------------------------------------
// Kernel 3: fused QKV GEMM. Block 256 threads (8 waves, 2x4), tile 64(M)x128(N),
// k-steps of 32, double-buffered async LDS staging.
// ---------------------------------------------------------------------------
__global__ __launch_bounds__(256)
void qkv_gemm_kernel(const bf16_t* __restrict__ Xb,  // [8192,1024] bf16
                     const bf16_t* __restrict__ Wt,  // [3][1024 n][1024 k] bf16
                     const float* __restrict__ bq, const float* __restrict__ bk,
                     const float* __restrict__ bv,
                     bf16_t* __restrict__ Qo, bf16_t* __restrict__ Ko,
                     bf16_t* __restrict__ Vo) {      // each [B,H,T,DK]
  __shared__ bf16_t Xs[2][64 * 32];    // 4 KB each
  __shared__ bf16_t Ws[2][128 * 32];   // 8 KB each
  const int tid   = threadIdx.x;
  const int lane  = tid & 31;
  const int wave  = tid >> 5;
  const int col16 = lane & 15;
  const int lh    = lane >> 4;
  const int waveM = wave >> 2;   // 0..1
  const int waveN = wave & 3;    // 0..3
  const int M0 = blockIdx.y * 64;
  const int ng = blockIdx.x * 128;          // global col in [0, 3072)
  const int mat = ng >> 10;                 // 0=Q 1=K 2=V (uniform per block)
  const int n0 = ng & 1023;
  const bf16_t* Wm = Wt + (size_t)mat * D_MODELC * D_MODELC;
  const float* bias = (mat == 0) ? bq : (mat == 1) ? bk : bv;
  bf16_t* out = (mat == 0) ? Qo : (mat == 1) ? Ko : Vo;
  const float scale = (mat == 0) ? 0.125f : 1.0f;   // 1/sqrt(DK) folded into Q

  auto stage = [&](int buf, int kk) {
    {  // X tile: 64 rows x 32 -> 256 b128 chunks, 1 per thread
      const int row = tid >> 2, sub = (tid & 3) * 8;
      cp16_g2s(&Xs[buf][row * 32 + sub],
               Xb + (size_t)(M0 + row) * D_MODELC + kk + sub);
    }
#pragma unroll
    for (int cch = 0; cch < 2; ++cch) {  // W tile: 128 rows x 32 -> 512 chunks
      const int idx = tid + cch * 256;
      const int row = idx >> 2, sub = (idx & 3) * 8;
      cp16_g2s(&Ws[buf][row * 32 + sub],
               Wm + (size_t)(n0 + row) * D_MODELC + kk + sub);
    }
  };

  const v8f vzero = {0.f, 0.f, 0.f, 0.f, 0.f, 0.f, 0.f, 0.f};
  v8f acc[2][2];
#pragma unroll
  for (int rt = 0; rt < 2; ++rt)
#pragma unroll
    for (int ct = 0; ct < 2; ++ct) acc[rt][ct] = vzero;

  stage(0, 0);
  const int NIT = D_MODELC / 32;
  for (int it = 0; it < NIT; ++it) {
    const int buf = it & 1;
    wait_async0();        // my chunks of `buf` have landed
    __syncthreads();      // everyone's chunks landed; prev-iter reads retired
    if (it + 1 < NIT) stage(buf ^ 1, (it + 1) * 32);  // overlap copy w/ compute

    v16bf afrag[2], bfrag[2];
#pragma unroll
    for (int rt = 0; rt < 2; ++rt)
      afrag[rt] = load_afrag(&Xs[buf][(waveM * 32 + rt * 16 + col16) * 32], lh * 8);
#pragma unroll
    for (int ct = 0; ct < 2; ++ct)
      bfrag[ct] = load_bfrag(&Ws[buf][(waveN * 32 + ct * 16 + col16) * 32], lh * 16);
#pragma unroll
    for (int rt = 0; rt < 2; ++rt)
#pragma unroll
      for (int ct = 0; ct < 2; ++ct)
        acc[rt][ct] = wmma_bf16(afrag[rt], bfrag[ct], acc[rt][ct]);
  }

  // Epilogue: C layout M = v + 8*lh, N = lane%16.
#pragma unroll
  for (int rt = 0; rt < 2; ++rt) {
#pragma unroll
    for (int ct = 0; ct < 2; ++ct) {
      const int n = n0 + waveN * 32 + ct * 16 + col16;
      const float bvv = bias[n];
      const int h = n >> 6, d = n & 63;
#pragma unroll
      for (int v = 0; v < 8; ++v) {
        const int m = M0 + waveM * 32 + rt * 16 + v + 8 * lh;
        const int b = m >> 11, t = m & 2047;
        out[(((size_t)(b * NHEADS + h) * T_SEQ + t) * DKC) + d] =
            (bf16_t)((acc[rt][ct][v] + bvv) * scale);
      }
    }
  }
}

// ---------------------------------------------------------------------------
// Kernel 4: causal flash attention. Grid (T/64, B*H), block 128 (4 waves),
// wave = 16 query rows. Double-buffered async K/V tiles (32 keys each).
// ---------------------------------------------------------------------------
__global__ __launch_bounds__(128)
void attn_kernel(const bf16_t* __restrict__ Q, const bf16_t* __restrict__ K,
                 const bf16_t* __restrict__ V, float* __restrict__ out) {
  __shared__ bf16_t Klds[2][32 * DKC];   // 4 KB each
  __shared__ bf16_t Vlds[2][32 * DKC];   // 4 KB each
  __shared__ bf16_t Plds[4][16 * 32];    // per-wave P staging

  const int lane  = threadIdx.x & 31;
  const int wave  = threadIdx.x >> 5;
  const int col16 = lane & 15;
  const int lh    = lane >> 4;
  const int bh    = blockIdx.y;
  const int tq0   = blockIdx.x * 64;
  const int q0    = tq0 + wave * 16;

  const bf16_t* Qb = Q + (size_t)bh * T_SEQ * DKC;
  const bf16_t* Kb = K + (size_t)bh * T_SEQ * DKC;
  const bf16_t* Vb = V + (size_t)bh * T_SEQ * DKC;

  auto stageKV = [&](int buf, int k0) {
#pragma unroll
    for (int cch = 0; cch < 2; ++cch) {  // 4 KB = 256 chunks; 128 threads x 2
      const int idx = threadIdx.x + cch * 128;
      cp16_g2s(&Klds[buf][idx * 8], Kb + (size_t)k0 * DKC + idx * 8);
      cp16_g2s(&Vlds[buf][idx * 8], Vb + (size_t)k0 * DKC + idx * 8);
    }
  };

  // Preload Q A-fragments (two 32-wide d-contraction steps).
  v16bf qf[2];
  {
    const bf16_t* qrow = Qb + (size_t)(q0 + col16) * DKC;
#pragma unroll
    for (int s = 0; s < 2; ++s) qf[s] = load_afrag(qrow + s * 32, lh * 8);
  }

  const v8f vzero = {0.f, 0.f, 0.f, 0.f, 0.f, 0.f, 0.f, 0.f};
  v8f oacc[4];
#pragma unroll
  for (int dt = 0; dt < 4; ++dt) oacc[dt] = vzero;
  float m_i[8], l_i[8];
#pragma unroll
  for (int v = 0; v < 8; ++v) { m_i[v] = -__builtin_inff(); l_i[v] = 0.f; }

  const int nkt = (tq0 >> 5) + 2;  // causal: keys < tq0 + 64
  stageKV(0, 0);
  for (int kt = 0; kt < nkt; ++kt) {
    const int buf = kt & 1;
    wait_async0();
    __syncthreads();
    if (kt + 1 < nkt) stageKV(buf ^ 1, (kt + 1) * 32);
    const int k0 = kt * 32;

    // Scores: two 16-key sub-tiles, 2 WMMAs each over d (Q pre-scaled).
    float sc[2][8];
#pragma unroll
    for (int j = 0; j < 2; ++j) {
      v8f s8 = vzero;
#pragma unroll
      for (int s = 0; s < 2; ++s) {
        const bf16_t* krow = &Klds[buf][(j * 16 + col16) * DKC + s * 32];
        s8 = wmma_bf16(qf[s], load_bfrag(krow, lh * 16), s8);
      }
#pragma unroll
      for (int v = 0; v < 8; ++v) {  // causal mask
        const int qrow = q0 + v + 8 * lh;
        const int key  = k0 + j * 16 + col16;
        sc[j][v] = (key <= qrow) ? s8[v] : -__builtin_inff();
      }
    }

    // Row-wise max across the 16 lanes holding each row.
    float tmax[8];
#pragma unroll
    for (int v = 0; v < 8; ++v) tmax[v] = fmaxf(sc[0][v], sc[1][v]);
#pragma unroll
    for (int off = 1; off < 16; off <<= 1)
#pragma unroll
      for (int v = 0; v < 8; ++v)
        tmax[v] = fmaxf(tmax[v], __shfl_xor(tmax[v], off, 32));

    float alpha[8];
#pragma unroll
    for (int v = 0; v < 8; ++v) {
      const float mnew = fmaxf(m_i[v], tmax[v]);
      alpha[v] = __expf(m_i[v] - mnew);  // exp(-inf)=0 on first tile
      m_i[v] = mnew;
    }

    float rsum[8];
#pragma unroll
    for (int v = 0; v < 8; ++v) rsum[v] = 0.f;
#pragma unroll
    for (int j = 0; j < 2; ++j) {
#pragma unroll
      for (int v = 0; v < 8; ++v) {
        const float p = __expf(sc[j][v] - m_i[v]);
        rsum[v] += p;
        Plds[wave][(v + 8 * lh) * 32 + j * 16 + col16] = (bf16_t)p;
      }
    }
#pragma unroll
    for (int off = 1; off < 16; off <<= 1)
#pragma unroll
      for (int v = 0; v < 8; ++v) rsum[v] += __shfl_xor(rsum[v], off, 32);
#pragma unroll
    for (int v = 0; v < 8; ++v) l_i[v] = l_i[v] * alpha[v] + rsum[v];
#pragma unroll
    for (int dt = 0; dt < 4; ++dt)
#pragma unroll
      for (int v = 0; v < 8; ++v) oacc[dt][v] *= alpha[v];

    __syncthreads();  // P visible wave-wide

    // P as A-fragment (16 rows x 32 keys).
    const v16bf pf = load_afrag(&Plds[wave][col16 * 32], lh * 8);

    // O += P(16x32) * V(32xDK): 4 WMMAs; V B-frags are strided over keys.
#pragma unroll
    for (int dt = 0; dt < 4; ++dt) {
      v16bf vfrag;
      const bf16_t* src = &Vlds[buf][(lh ? 16 : 0) * DKC + dt * 16 + col16];
#pragma unroll
      for (int i = 0; i < 8; ++i) {
        vfrag[2 * i]     = src[(size_t)(2 * i) * DKC];
        vfrag[2 * i + 1] = src[(size_t)(2 * i + 1) * DKC];
      }
      oacc[dt] = wmma_bf16(pf, vfrag, oacc[dt]);
    }
  }

  // Normalize and write fp32 [B, T, D_MODEL].
  const int b = bh >> 4, h = bh & 15;
#pragma unroll
  for (int dt = 0; dt < 4; ++dt) {
#pragma unroll
    for (int v = 0; v < 8; ++v) {
      const int qrow = q0 + v + 8 * lh;
      const int d = dt * 16 + col16;
      out[((size_t)(b * T_SEQ + qrow)) * D_MODELC + h * DKC + d] =
          oacc[dt][v] / l_i[v];
    }
  }
}

// ---------------------------------------------------------------------------
extern "C" void kernel_launch(void* const* d_in, const int* in_sizes, int n_in,
                              void* d_out, int out_size, void* d_ws, size_t ws_size,
                              hipStream_t stream) {
  const float* x  = (const float*)d_in[0];
  const float* Wq = (const float*)d_in[1];
  const float* bq = (const float*)d_in[2];
  const float* Wk = (const float*)d_in[3];
  const float* bk = (const float*)d_in[4];
  const float* Wv = (const float*)d_in[5];
  const float* bv = (const float*)d_in[6];

  // Workspace layout (bf16 elements):
  //   xb [8192*1024] | Wt [3*1024*1024] | Q | K | V (each 8M) -> ~70 MB total
  const size_t nX  = (size_t)BATCH * T_SEQ * D_MODELC;       // 8M
  const size_t nW  = (size_t)3 * D_MODELC * D_MODELC;        // 3M
  const size_t nP  = (size_t)BATCH * NHEADS * T_SEQ * DKC;   // 8M
  bf16_t* xb = (bf16_t*)d_ws;
  bf16_t* Wt = xb + nX;
  bf16_t* Qw = Wt + nW;
  bf16_t* Kw = Qw + nP;
  bf16_t* Vw = Kw + nP;

  convert_x_kernel<<<dim3((unsigned)(nX / 4 / 256)), 256, 0, stream>>>(
      x, xb, (int)(nX / 4));
  convert_wt_kernel<<<dim3(D_MODELC / 32, D_MODELC / 32, 3), 256, 0, stream>>>(
      Wq, Wk, Wv, Wt);

  dim3 g1(3 * D_MODELC / 128, (BATCH * T_SEQ) / 64);  // (24, 128)
  qkv_gemm_kernel<<<g1, 256, 0, stream>>>(xb, Wt, bq, bk, bv, Qw, Kw, Vw);

  dim3 g2(T_SEQ / 64, BATCH * NHEADS);                // (32, 64)
  attn_kernel<<<g2, 128, 0, stream>>>(Qw, Kw, Vw, (float*)d_out);
}
